// Seq2SeqLSTM_36799279792942
// MI455X (gfx1250) — matrix operand order repaired and checked by reading, not compiled, not run on hardware
//
#include <hip/hip_runtime.h>
#include <hip/hip_bf16.h>

// ---------------------------------------------------------------------------
// Seq2Seq LSTM for MI455X (gfx1250): bf16 WMMA recurrent GEMMs with
// double-buffered async-to-LDS tile staging, fused gate pointwise kernels.
// ---------------------------------------------------------------------------
#define BS 64
#define SL 256
#define FL 256
#define SI 512
#define SH 1024
#define NG (4 * SH)   // 4096 fused gate columns

// GEMM tiling: 64(M) x 128(N) per workgroup, K-step 64, double-buffered LDS.
#define KSTEP  64
#define PITCH  144                       // 128B row + 16B pad (bank-conflict free, 16B aligned)
#define A_BYTES (64 * PITCH)             // 9216
#define B_BYTES (128 * PITCH)            // 18432
#define STAGE_BYTES (A_BYTES + B_BYTES)  // 27648 per stage, x2 stages = 55296 B

typedef __attribute__((ext_vector_type(16))) __bf16 v16bf;
typedef __attribute__((ext_vector_type(8)))  float  v8f;
typedef __attribute__((ext_vector_type(4)))  int    i4;
typedef __attribute__((address_space(3))) unsigned char as3_uchar;

// ---- helpers --------------------------------------------------------------
__device__ __forceinline__ unsigned short f2bf(float f) {
    union { float f; unsigned u; } x; x.f = f;
    unsigned u = x.u;
    unsigned r = u + 0x7FFFu + ((u >> 16) & 1u);   // round-to-nearest-even
    return (unsigned short)(r >> 16);
}
__device__ __forceinline__ float sigmoidf_(float x) {
    return 1.0f / (1.0f + __expf(-x));
}

// Async global -> LDS 16B copy (CDNA5 ASYNCcnt path, ISA 08 section 4).
__device__ __forceinline__ void async_b128(void* lds, const void* gptr) {
    unsigned loff = (unsigned)(unsigned long long)(as3_uchar*)lds;
    asm volatile("global_load_async_to_lds_b128 %0, %1, off"
                 :: "v"(loff), "v"(gptr)
                 : "memory");
}
__device__ __forceinline__ void wait_async_le6() {
    asm volatile("s_wait_asynccnt 0x6" ::: "memory");
}
__device__ __forceinline__ void wait_async_0() {
    asm volatile("s_wait_asynccnt 0x0" ::: "memory");
}

// ---- fused-gate GEMM kernel ------------------------------------------------
// out[M x N] = A1[M x K1] * B1^T (+ A2[M x K2] * B2^T) + bias[N]
// mode 0: store z rows (M=64, N=4096, row-major)
// mode 1: final projection: sigmoid + scatter to out[b][t][o] (M=FL*BS, N=SI)
//
// WMMA fragment layouts per ISA 7.12.2:
//  A 16x32 bf16: lanes 0-15 row=lane, K {0..7,16..23}; lanes 16-31 K {8..15,24..31}
//  B 32x16 bf16: lane col = lane&15; lanes 0-15 K {0..7,16..23}; lanes 16-31 K {8..15,24..31}
//  C/D v8f: VGPR j = row j (lanes 0-15) / row j+8 (lanes 16-31), col = lane&15
__global__ __launch_bounds__(256) void lstm_gemm_kernel(
        const unsigned short* __restrict__ A1, int lda1,
        const unsigned short* __restrict__ B1, int K1,
        const unsigned short* __restrict__ A2, int lda2,
        const unsigned short* __restrict__ B2, int K2,
        const float* __restrict__ bias, float* __restrict__ out, int N, int mode) {
    __shared__ __align__(16) unsigned char smem[2 * STAGE_BYTES];

    const int tid  = threadIdx.x;
    const int lane = tid & 31;
    const int wave = tid >> 5;
    const int wm = wave & 3;       // M subtile 0..3
    const int wn = wave >> 2;      // N subtile 0..1
    const int mBlock = blockIdx.y * 64;
    const int nBlock = blockIdx.x * 128;
    const int m0 = mBlock + wm * 16;
    const int n0 = nBlock + wn * 64;

    const int s1 = K1 >> 6;                          // stages in phase 1
    const int S  = s1 + (A2 ? (K2 >> 6) : 0);        // total K stages

    v8f acc[4] = {};

    // Issue one stage's cooperative async tile loads (6 x b128 per thread).
    auto issue = [&](int s) {
        const unsigned short* Ag; const unsigned short* Bg; int lda, K, kb;
        if (s < s1) { Ag = A1; Bg = B1; lda = lda1; K = K1; kb = s << 6; }
        else        { Ag = A2; Bg = B2; lda = lda2; K = K2; kb = (s - s1) << 6; }
        unsigned char* bufA = smem + (s & 1) * STAGE_BYTES;
        unsigned char* bufB = bufA + A_BYTES;
#pragma unroll
        for (int i = 0; i < 2; ++i) {                // A tile: 64 rows x 128B
            const int cid = tid + i * 256;
            const int row = cid >> 3, cc = cid & 7;
            async_b128(bufA + row * PITCH + cc * 16,
                       Ag + (long)(mBlock + row) * lda + kb + cc * 8);
        }
#pragma unroll
        for (int i = 0; i < 4; ++i) {                // B tile: 128 rows x 128B
            const int cid = tid + i * 256;
            const int row = cid >> 3, cc = cid & 7;
            async_b128(bufB + row * PITCH + cc * 16,
                       Bg + (long)(nBlock + row) * K + kb + cc * 8);
        }
    };

    // Consume one stage from LDS. All 20 ds_load_b128 fragment reads are
    // hoisted ahead of the 8 back-to-back WMMAs so the LDS latency is paid
    // once per stage, not once per WMMA.
    union Frag { v16bf v; i4 q[2]; };
    auto compute = [&](int s) {
        const unsigned char* bufA = smem + (s & 1) * STAGE_BYTES;
        const unsigned char* bufB = bufA + A_BYTES;
        const unsigned char* fa  = bufA + (wm * 16 + (lane & 15)) * PITCH + ((lane >> 4) << 4);
        const unsigned char* fb0 = bufB + (wn * 64 + (lane & 15)) * PITCH + ((lane >> 4) << 4);

        Frag a[2];
        Frag b[2][4];
#pragma unroll
        for (int ks = 0; ks < 2; ++ks) {             // 32 K elements = 64 bytes each
            const int kb = ks * 64;
            a[ks].q[0] = *(const i4*)(fa + kb);
            a[ks].q[1] = *(const i4*)(fa + kb + 32);
#pragma unroll
            for (int t = 0; t < 4; ++t) {
                const unsigned char* fb = fb0 + t * 16 * PITCH + kb;
                b[ks][t].q[0] = *(const i4*)(fb);
                b[ks][t].q[1] = *(const i4*)(fb + 32);
            }
        }
#pragma unroll
        for (int ks = 0; ks < 2; ++ks)
#pragma unroll
            for (int t = 0; t < 4; ++t)
                acc[t] = __builtin_amdgcn_wmma_f32_16x16x32_bf16(
                    false, a[ks].v, false, b[ks][t].v, (short)0, acc[t], false, false);
    };

    // Software pipeline: stage s+1 in flight while stage s computes.
    issue(0);
    for (int s = 0; s < S; ++s) {
        if (s + 1 < S) { issue(s + 1); wait_async_le6(); }
        else           { wait_async_0(); }
        __syncthreads();
        compute(s);
        __syncthreads();
    }

    // Epilogue.
    const int rowBase = m0 + ((lane >> 4) << 3);
#pragma unroll
    for (int t = 0; t < 4; ++t) {
        const int col = n0 + t * 16 + (lane & 15);
        const float bv = bias[col];
#pragma unroll
        for (int j = 0; j < 8; ++j) {
            const int m = rowBase + j;
            const float v = acc[t][j] + bv;
            if (mode == 0) {
                out[(long)m * N + col] = v;
            } else {
                const int b = m & (BS - 1);
                const int tt = m >> 6;
                out[((long)b * FL + tt) * (long)N + col] = sigmoidf_(v);
            }
        }
    }
}

// ---- pointwise LSTM cell update -------------------------------------------
// z: [64][4096] (bias already added); gate order f,i,g,o. Updates c (fp32),
// writes h as bf16 (recurrent operand), optionally mirrors into ys buffer.
__global__ __launch_bounds__(256) void lstm_gates_kernel(
        const float* __restrict__ Z, float* __restrict__ c,
        unsigned short* __restrict__ hb, unsigned short* __restrict__ hb2) {
    const int idx = blockIdx.x * 256 + threadIdx.x;       // 0..65535
    const int b = idx >> 10, hh = idx & (SH - 1);
    const float* zr = Z + (long)b * NG;
    const float f = sigmoidf_(zr[hh]);
    const float i = sigmoidf_(zr[SH + hh]);
    const float g = tanhf(zr[2 * SH + hh]);
    const float o = sigmoidf_(zr[3 * SH + hh]);
    const float cn = f * c[idx] + i * g;
    const float hn = o * tanhf(cn);
    c[idx] = cn;
    const unsigned short hv = f2bf(hn);
    hb[idx] = hv;
    if (hb2) hb2[idx] = hv;
}

// ---- conversion / copy ----------------------------------------------------
__global__ void f32_to_bf16_kernel(const float* __restrict__ s,
                                   unsigned short* __restrict__ d, int n) {
    int i = blockIdx.x * 256 + threadIdx.x;
    if (i < n) d[i] = f2bf(s[i]);
}
__global__ void copy_f32_kernel(const float* __restrict__ s,
                                float* __restrict__ d, int n) {
    int i = blockIdx.x * 256 + threadIdx.x;
    if (i < n) d[i] = s[i];
}

// ---------------------------------------------------------------------------
extern "C" void kernel_launch(void* const* d_in, const int* in_sizes, int n_in,
                              void* d_out, int out_size, void* d_ws, size_t ws_size,
                              hipStream_t stream) {
    const float* x       = (const float*)d_in[0];
    const float* h0      = (const float*)d_in[1];
    const float* c0      = (const float*)d_in[2];
    const float* enc0_Wu = (const float*)d_in[3];
    const float* enc0_bu = (const float*)d_in[4];
    const float* enc0_Wv = (const float*)d_in[5];
    const float* enc1_Wu = (const float*)d_in[6];
    const float* enc1_bu = (const float*)d_in[7];
    const float* enc1_Wv = (const float*)d_in[8];
    const float* dec0_Wu = (const float*)d_in[9];
    const float* dec0_bu = (const float*)d_in[10];
    const float* dec0_Wv = (const float*)d_in[11];
    const float* dec1_Wu = (const float*)d_in[12];
    const float* dec1_bu = (const float*)d_in[13];
    const float* dec1_Wv = (const float*)d_in[14];
    const float* fin_W   = (const float*)d_in[15];
    const float* fin_b   = (const float*)d_in[16];
    (void)in_sizes; (void)n_in; (void)out_size; (void)ws_size;

    // ---- workspace carve-up (bf16 weights L2-resident: ~61 MB) ------------
    char* ws = (char*)d_ws;
    size_t off = 0;
    auto alloc = [&](size_t bytes) -> void* {
        void* p = ws + off;
        off = (off + bytes + 255) & ~(size_t)255;
        return p;
    };
    unsigned short* wE0u = (unsigned short*)alloc((size_t)NG * SI * 2);
    unsigned short* wE0v = (unsigned short*)alloc((size_t)NG * SH * 2);
    unsigned short* wE1u = (unsigned short*)alloc((size_t)NG * SH * 2);
    unsigned short* wE1v = (unsigned short*)alloc((size_t)NG * SH * 2);
    unsigned short* wD0u = (unsigned short*)alloc((size_t)NG * SH * 2);
    unsigned short* wD0v = (unsigned short*)alloc((size_t)NG * SH * 2);
    unsigned short* wD1u = (unsigned short*)alloc((size_t)NG * SH * 2);
    unsigned short* wD1v = (unsigned short*)alloc((size_t)NG * SH * 2);
    unsigned short* wFin = (unsigned short*)alloc((size_t)SI * SH * 2);
    unsigned short* xb   = (unsigned short*)alloc((size_t)BS * SL * SI * 2);
    unsigned short* h0b  = (unsigned short*)alloc((size_t)BS * SH * 2);
    unsigned short* h1b  = (unsigned short*)alloc((size_t)BS * SH * 2);
    float*          c0f  = (float*)alloc((size_t)BS * SH * 4);
    float*          c1f  = (float*)alloc((size_t)BS * SH * 4);
    float*          zbuf = (float*)alloc((size_t)BS * NG * 4);
    unsigned short* ysb  = (unsigned short*)alloc((size_t)FL * BS * SH * 2);

    auto conv = [&](const float* s, unsigned short* d, long n) {
        f32_to_bf16_kernel<<<dim3((unsigned)((n + 255) / 256)), 256, 0, stream>>>(s, d, (int)n);
    };

    // ---- one-time precision conversion ------------------------------------
    conv(x, xb, (long)BS * SL * SI);
    conv(h0, h0b, (long)BS * SH);
    conv(h0 + (long)BS * SH, h1b, (long)BS * SH);
    copy_f32_kernel<<<dim3(BS * SH / 256), 256, 0, stream>>>(c0, c0f, BS * SH);
    copy_f32_kernel<<<dim3(BS * SH / 256), 256, 0, stream>>>(c0 + (long)BS * SH, c1f, BS * SH);
    conv(enc0_Wu, wE0u, (long)NG * SI);
    conv(enc0_Wv, wE0v, (long)NG * SH);
    conv(enc1_Wu, wE1u, (long)NG * SH);
    conv(enc1_Wv, wE1v, (long)NG * SH);
    conv(dec0_Wu, wD0u, (long)NG * SH);
    conv(dec0_Wv, wD0v, (long)NG * SH);
    conv(dec1_Wu, wD1u, (long)NG * SH);
    conv(dec1_Wv, wD1v, (long)NG * SH);
    conv(fin_W,   wFin, (long)SI * SH);

    const dim3 gemmGrid(NG / 128, 1);
    const dim3 gateGrid(BS * SH / 256);
    unsigned short* const noUS = nullptr;

    // ---- encoder: 256 sequential steps ------------------------------------
    for (int t = 0; t < SL; ++t) {
        lstm_gemm_kernel<<<gemmGrid, 256, 0, stream>>>(
            xb + (long)t * SI, SL * SI, wE0u, SI,
            h0b, SH, wE0v, SH, enc0_bu, zbuf, NG, 0);
        lstm_gates_kernel<<<gateGrid, 256, 0, stream>>>(zbuf, c0f, h0b, noUS);
        lstm_gemm_kernel<<<gemmGrid, 256, 0, stream>>>(
            h0b, SH, wE1u, SH,
            h1b, SH, wE1v, SH, enc1_bu, zbuf, NG, 0);
        lstm_gates_kernel<<<gateGrid, 256, 0, stream>>>(zbuf, c1f, h1b, noUS);
    }

    // ---- decoder: 256 sequential steps (states carried in-place) ----------
    for (int d = 0; d < FL; ++d) {
        // layer0: z = st@Wu + ha@Wv ; st = previous h1 (enc-final at d==0)
        lstm_gemm_kernel<<<gemmGrid, 256, 0, stream>>>(
            h1b, SH, wD0u, SH,
            h0b, SH, wD0v, SH, dec0_bu, zbuf, NG, 0);
        lstm_gates_kernel<<<gateGrid, 256, 0, stream>>>(zbuf, c0f, h0b, noUS);
        // layer1: z = ha_new@Wu + hb_prev@Wv ; emit ys[d]
        lstm_gemm_kernel<<<gemmGrid, 256, 0, stream>>>(
            h0b, SH, wD1u, SH,
            h1b, SH, wD1v, SH, dec1_bu, zbuf, NG, 0);
        lstm_gates_kernel<<<gateGrid, 256, 0, stream>>>(
            zbuf, c1f, h1b, ysb + (long)d * BS * SH);
    }

    // ---- final projection: [FL*BS,1024] x [512,1024]^T, fused sigmoid -----
    lstm_gemm_kernel<<<dim3(SI / 128, (FL * BS) / 64), 256, 0, stream>>>(
        ysb, SH, wFin, SH,
        noUS, 0, noUS, 0, fin_b, (float*)d_out, SI, 1);
}
